// FlashSVDLlamaBlock_26980984553791
// MI455X (gfx1250) — matrix operand order repaired and checked by reading, not compiled, not run on hardware
//
#include <hip/hip_runtime.h>

// ---------------------------------------------------------------------------
// FlashSVD LLaMA block for gfx1250 (MI455X, wave32, WMMA + TDM).
// All GEMMs -> v_wmma_f32_16x16x32_bf16 (bf16 operands, f32 accumulate).
// fp32 weights are converted to bf16 during the global->LDS staging step.
// bf16 activation tiles are staged via the Tensor Data Mover
// (tensor_load_to_lds) with hardware pad insertion matching the LDS layout.
// Problem dims: T=1024, D=4096, H=32, HK=8, DH=128, RQ/RKV=64, RO/RFF=1024,
// INTER=14336. All dims are multiples of the 128x128x32 tile -> no edge code.
// Workspace use: ~52 MB.
// ---------------------------------------------------------------------------

typedef unsigned short u16;
typedef __attribute__((ext_vector_type(16))) __bf16 v16bf;
typedef __attribute__((ext_vector_type(8)))  float  v8f;

union Frag { v16bf v; uint4 q[2]; };

static __device__ __forceinline__ u16 f2bf(float f) {
  union { float f; unsigned u; } v; v.f = f;
  unsigned r = v.u + 0x7FFFu + ((v.u >> 16) & 1u);   // RNE
  return (u16)(r >> 16);
}
static __device__ __forceinline__ float bf2f(u16 h) {
  union { unsigned u; float f; } v; v.u = ((unsigned)h) << 16;
  return v.f;
}

// ---------------------------------------------------------------------------
// Tensor Data Mover: 2D bf16 tile global->LDS with pad insertion.
// D# layout per CDNA5 ISA ch.8 (group0 128b, group1 256b, groups2/3 zero).
// tensor dims == tile dims (no OOB clipping wanted), dim0 stride = row stride.
// Issued by one wave; completion via s_wait_tensorcnt + workgroup barrier.
// This toolchain exposes the 6-arg builtin:
//   (uint32x4 g0, int32x8 g1, int32x4 g2, int32x4 g3, int32x8 extra, i32 cpol)
// ---------------------------------------------------------------------------
#if defined(__has_builtin)
#if __has_builtin(__builtin_amdgcn_tensor_load_to_lds) && \
    __has_builtin(__builtin_amdgcn_s_wait_tensorcnt)
#define HAVE_TDM 1
#endif
#endif
#ifndef HAVE_TDM
#define HAVE_TDM 0
#endif

#if HAVE_TDM
typedef __attribute__((ext_vector_type(4))) unsigned int u32x4;
typedef __attribute__((ext_vector_type(8))) int i32x8;
typedef __attribute__((ext_vector_type(4))) int i32x4;

static __device__ __forceinline__ unsigned lds_addr_of(const void* p) {
  // generic->LDS: low 32 bits of the flat address are the LDS byte offset
  return (unsigned)(unsigned long long)(size_t)p;
}

// padIntervalCode: dwords before pad = 2<<code ; padAmountCode: pad dwords - 1
static __device__ __forceinline__ void tdm_load_2d_bf16(
    unsigned ldsAddr, const void* gptr, unsigned tileW, unsigned tileH,
    unsigned long long rowStrideElems, unsigned padIntervalCode,
    unsigned padAmountCode) {
  unsigned long long ga = (unsigned long long)(size_t)gptr;
  u32x4 g0;
  g0[0] = 0x1u;                                  // count=1, user descriptor
  g0[1] = ldsAddr;                               // lds_addr (bytes)
  g0[2] = (unsigned)ga;                          // global_addr[31:0]
  g0[3] = (unsigned)((ga >> 32) & 0x01FFFFFFu)   // global_addr[56:32]
          | (2u << 30);                          // type = 2 ("image")
  i32x8 g1;
  g1[0] = (int)((1u << 16)                       // data_size = 2 bytes
                | (1u << 20)                     // pad_enable
                | (padIntervalCode << 22) | (padAmountCode << 25));
  g1[1] = (int)(tileW << 16);                    // tensor_dim0[15:0] @ bits 63:48
  g1[2] = (int)((tileW >> 16) | (tileH << 16));  // tensor_dim0 hi | tensor_dim1 lo
  g1[3] = (int)((tileH >> 16) | (tileW << 16));  // tensor_dim1 hi | tile_dim0
  g1[4] = (int)(tileH & 0xFFFFu);                // tile_dim1 (tile_dim2 = 0)
  g1[5] = (int)(unsigned)(rowStrideElems & 0xFFFFFFFFull);     // dim0 stride lo
  g1[6] = (int)(unsigned)((rowStrideElems >> 32) & 0xFFFFull); // stride hi
  g1[7] = 0;
  i32x4 z4; z4[0] = z4[1] = z4[2] = z4[3] = 0;
  i32x8 z8;
  for (int i = 0; i < 8; ++i) z8[i] = 0;
  __builtin_amdgcn_tensor_load_to_lds(g0, g1, z4, z4, z8, 0);
}
#endif

// ---------------------------------------------------------------------------
// Generic WMMA GEMM:  C[M,N] (+)= A[M,K] * B[N,K]^T
//   AF32/BF32: operand stored as fp32 (converted to bf16 while staging to LDS)
//   CADD:      false -> C is bf16 (overwrite); true -> C is fp32 (read-add-write)
// Block tile 128x128, BK=32, 256 threads (8 waves), wave tile 32x64 (2x4 wmma).
// bf16 A tiles staged by TDM (wave 0), fp32 operands converted by VALU.
// grid = (N/128, M/128, batch)
// ---------------------------------------------------------------------------
template <bool AF32, bool BF32, bool CADD>
__global__ __launch_bounds__(256) void gemm_wmma_kernel(
    const void* __restrict__ Ap, int lda, long long sA,
    const void* __restrict__ Bp, int ldb, long long sB,
    void* __restrict__ Cp, int ldc, long long sC,
    int K) {
  __shared__ __align__(16) u16 As[128][40];   // 32 data + 8 pad halves per row
  __shared__ __align__(16) u16 Bs[128][40];

  const int tid  = threadIdx.x;
  const int lane = tid & 31;
  const int wave = tid >> 5;
  const int wm   = (wave >> 1) * 32;   // wave row offset in block tile
  const int wn   = (wave & 1) * 64;    // wave col offset in block tile
  const int mBase = blockIdx.y * 128;
  const int nBase = blockIdx.x * 128;
  const long long bz = blockIdx.z;

  const float* Af = (const float*)Ap + sA * bz;
  const u16*   Ab = (const u16*)Ap   + sA * bz;
  const float* Bf = (const float*)Bp + sB * bz;
  const u16*   Bb = (const u16*)Bp   + sB * bz;

  v8f acc[2][4];
  for (int i = 0; i < 2; ++i)
    for (int j = 0; j < 4; ++j)
      for (int v = 0; v < 8; ++v) acc[i][j][v] = 0.f;

  const int cl = lane & 15;
  const int hi = (lane >= 16) ? 1 : 0;
  const int ak = hi ? 8 : 0;    // A k-run start (halves)
  const int bk = hi ? 8 : 0;    // B k-run start (halves)

  for (int kt = 0; kt < K; kt += 32) {
    // ---- stage A tile (128x32) ----
    if constexpr (AF32) {
      for (int i = 0; i < 2; ++i) {
        int slot = tid + i * 256;            // 0..511
        int r = slot >> 2, c8 = (slot & 3) * 8;
        const float* p = Af + (size_t)(mBase + r) * lda + kt + c8;
        float4 f0 = *(const float4*)p;
        float4 f1 = *(const float4*)(p + 4);
        union { u16 us[8]; uint4 q; } t;
        t.us[0] = f2bf(f0.x); t.us[1] = f2bf(f0.y); t.us[2] = f2bf(f0.z); t.us[3] = f2bf(f0.w);
        t.us[4] = f2bf(f1.x); t.us[5] = f2bf(f1.y); t.us[6] = f2bf(f1.z); t.us[7] = f2bf(f1.w);
        *(uint4*)&As[r][c8] = t.q;
      }
    } else {
#if HAVE_TDM
      if (wave == 0)
        tdm_load_2d_bf16(lds_addr_of(&As[0][0]),
                         Ab + (size_t)mBase * lda + kt, 32u, 128u,
                         (unsigned long long)lda, /*16dw*/ 3u, /*4dw*/ 3u);
#else
      for (int i = 0; i < 2; ++i) {
        int slot = tid + i * 256;
        int r = slot >> 2, c8 = (slot & 3) * 8;
        *(uint4*)&As[r][c8] =
            *(const uint4*)(Ab + (size_t)(mBase + r) * lda + kt + c8);
      }
#endif
    }
    // ---- stage B tile (128x32) ----
    for (int i = 0; i < 2; ++i) {
      int slot = tid + i * 256;
      int r = slot >> 2, c8 = (slot & 3) * 8;
      if (BF32) {
        const float* p = Bf + (size_t)(nBase + r) * ldb + kt + c8;
        float4 f0 = *(const float4*)p;
        float4 f1 = *(const float4*)(p + 4);
        if (kt + 32 < K) __builtin_prefetch(p + 32, 0, 1);  // next K slice
        union { u16 us[8]; uint4 q; } t;
        t.us[0] = f2bf(f0.x); t.us[1] = f2bf(f0.y); t.us[2] = f2bf(f0.z); t.us[3] = f2bf(f0.w);
        t.us[4] = f2bf(f1.x); t.us[5] = f2bf(f1.y); t.us[6] = f2bf(f1.z); t.us[7] = f2bf(f1.w);
        *(uint4*)&Bs[r][c8] = t.q;
      } else {
        *(uint4*)&Bs[r][c8] =
            *(const uint4*)(Bb + (size_t)(nBase + r) * ldb + kt + c8);
      }
    }
#if HAVE_TDM
    if constexpr (!AF32) {
      if (wave == 0) __builtin_amdgcn_s_wait_tensorcnt(0);
    }
#endif
    __syncthreads();

    // ---- fragments + WMMA ----
    Frag aF[2], bF[4];
    for (int i = 0; i < 2; ++i) {
      const u16* row = &As[wm + i * 16 + cl][0];
      aF[i].q[0] = *(const uint4*)(row + ak);
      aF[i].q[1] = *(const uint4*)(row + ak + 16);
    }
    for (int j = 0; j < 4; ++j) {
      const u16* row = &Bs[wn + j * 16 + cl][0];
      bF[j].q[0] = *(const uint4*)(row + bk);
      bF[j].q[1] = *(const uint4*)(row + bk + 16);
    }
    for (int i = 0; i < 2; ++i)
      for (int j = 0; j < 4; ++j)
        acc[i][j] = __builtin_amdgcn_wmma_f32_16x16x32_bf16(
            false, aF[i].v, false, bF[j].v, (short)0, acc[i][j], false, false);
    __syncthreads();
  }

  // ---- epilogue ----
  u16*   Cb = (u16*)Cp   + sC * bz;
  float* Cf = (float*)Cp + sC * bz;
  for (int i = 0; i < 2; ++i)
    for (int j = 0; j < 4; ++j) {
      int col  = nBase + wn + j * 16 + cl;
      int rowb = mBase + wm + i * 16 + (hi ? 8 : 0);
      for (int v = 0; v < 8; ++v) {
        size_t idx = (size_t)(rowb + v) * ldc + col;
        if (CADD) Cf[idx] += acc[i][j][v];
        else      Cb[idx]  = f2bf(acc[i][j][v]);
      }
    }
}

// ---------------------------------------------------------------------------
// RMSNorm: out(bf16)[row] = x[row] * rsqrt(mean(x^2)+eps) * w
// ---------------------------------------------------------------------------
__global__ __launch_bounds__(256) void rmsnorm_kernel(
    const float* __restrict__ x, const float* __restrict__ w,
    u16* __restrict__ out, int D) {
  const int tid = threadIdx.x, lane = tid & 31, wave = tid >> 5;
  const float* xr = x + (size_t)blockIdx.x * D;
  float s = 0.f;
  for (int i = tid; i < D; i += 256) { float v = xr[i]; s += v * v; }
  for (int o = 16; o > 0; o >>= 1) s += __shfl_xor(s, o, 32);
  __shared__ float red[8];
  __shared__ float scl;
  if (lane == 0) red[wave] = s;
  __syncthreads();
  if (tid == 0) {
    float t = 0.f;
    for (int i = 0; i < 8; ++i) t += red[i];
    scl = rsqrtf(t / (float)D + 1e-5f);
  }
  __syncthreads();
  float sc = scl;
  u16* orow = out + (size_t)blockIdx.x * D;
  for (int i = tid; i < D; i += 256) orow[i] = f2bf(xr[i] * sc * w[i]);
}

// ---------------------------------------------------------------------------
// RoPE (interleaved-angle / rotate_half), in-place on bf16 q [T,4096], k [T,1024]
// ---------------------------------------------------------------------------
__global__ __launch_bounds__(256) void rope_kernel(u16* __restrict__ q,
                                                   u16* __restrict__ k) {
  const int t = blockIdx.x;
  const float tf = (float)t;
  for (int p = threadIdx.x; p < 2048 + 512; p += 256) {
    u16* base;
    int d;
    if (p < 2048) { int h = p >> 6; d = p & 63; base = q + (size_t)t * 4096 + h * 128; }
    else          { int pp = p - 2048; int h = pp >> 6; d = pp & 63; base = k + (size_t)t * 1024 + h * 128; }
    float x0 = bf2f(base[d]);
    float x1 = bf2f(base[d + 64]);
    float e0 = (float)(d & ~1);  // 2*(d/2)
    float f0 = __powf(10000.f, -e0 / 128.f);
    float f1 = __powf(10000.f, -(e0 + 64.f) / 128.f);
    float c0, s0, c1, s1;
    __sincosf(tf * f0, &s0, &c0);
    __sincosf(tf * f1, &s1, &c1);
    float o0 = x0 * c0 - x1 * s0;   // rotate_half: -x[d+64]
    float o1 = x1 * c1 + x0 * s1;   //               x[d]
    base[d]      = f2bf(o0);
    base[d + 64] = f2bf(o1);
  }
}

// ---------------------------------------------------------------------------
// Flash attention (causal, GQA 4:1). Br=64 rows/block, Bc=32 keys/iter.
// grid = (T/64, H), 256 threads. Q/K tiles via TDM, softmax fp32.
// ---------------------------------------------------------------------------
__global__ __launch_bounds__(256) void flash_kernel(
    const u16* __restrict__ qb, const u16* __restrict__ kb,
    const u16* __restrict__ vb, u16* __restrict__ attn) {
  const int tid = threadIdx.x, lane = tid & 31, wave = tid >> 5;
  const int ib = blockIdx.x, h = blockIdx.y, kvh = h >> 2;

  __shared__ __align__(16) u16  Qs[64][136];   // 128 data + 8 pad halves
  __shared__ __align__(16) u16  Ks[32][136];
  __shared__ __align__(16) u16  Vt[128][40];   // V transposed: [dh][s]
  __shared__ float Ss[64][36];
  __shared__ __align__(16) u16  Ps[64][40];
  __shared__ float mS[64], lS[64], aS[64];

  // load Q tile (64 x 128)
#if HAVE_TDM
  if (wave == 0)
    tdm_load_2d_bf16(lds_addr_of(&Qs[0][0]),
                     qb + (size_t)(ib * 64) * 4096 + h * 128, 128u, 64u,
                     4096ull, /*64dw*/ 5u, /*4dw*/ 3u);
#else
  for (int i = 0; i < 4; ++i) {
    int slot = tid + i * 256;                       // 0..1023
    int r = slot >> 4, c8 = (slot & 15) * 8;
    *(uint4*)&Qs[r][c8] =
        *(const uint4*)(qb + (size_t)(ib * 64 + r) * 4096 + h * 128 + c8);
  }
#endif
  if (tid < 64) { mS[tid] = -1e30f; lS[tid] = 0.f; }

  const int cl = lane & 15;
  const int hi = (lane >= 16) ? 1 : 0;
  const int ak = hi ? 8 : 0;
  const int bk = hi ? 8 : 0;
  const int om = (wave >> 1) * 16;   // row tile (also S row tile)
  const int on = (wave & 1) * 64;    // O col base
  const int sn = (wave & 1) * 16;    // S col tile

  v8f o[4];
  for (int j = 0; j < 4; ++j)
    for (int v = 0; v < 8; ++v) o[j][v] = 0.f;

  const float scale = 0.0883883476483184f;  // 1/sqrt(128)
  const int nj = 2 * ib + 2;
  __syncthreads();

  for (int jb = 0; jb < nj; ++jb) {
    // stage K (32x128) via TDM, V^T (128x32) via VALU transpose
#if HAVE_TDM
    if (wave == 0)
      tdm_load_2d_bf16(lds_addr_of(&Ks[0][0]),
                       kb + (size_t)(jb * 32) * 1024 + kvh * 128, 128u, 32u,
                       1024ull, /*64dw*/ 5u, /*4dw*/ 3u);
#else
    for (int i = 0; i < 2; ++i) {
      int slot = tid + i * 256;                     // 0..511
      int r = slot >> 4, c8 = (slot & 15) * 8;
      *(uint4*)&Ks[r][c8] =
          *(const uint4*)(kb + (size_t)(jb * 32 + r) * 1024 + kvh * 128 + c8);
    }
#endif
    for (int i = 0; i < 2; ++i) {
      int slot = tid + i * 256;
      int s = slot >> 4, d8 = (slot & 15) * 8;
      uint4 vv = *(const uint4*)(vb + (size_t)(jb * 32 + s) * 1024 + kvh * 128 + d8);
      const u16* hp = (const u16*)&vv;
      for (int e = 0; e < 8; ++e) Vt[d8 + e][s] = hp[e];
    }
#if HAVE_TDM
    if (wave == 0) __builtin_amdgcn_s_wait_tensorcnt(0);  // covers Q on jb==0 too
#endif
    __syncthreads();

    // S = Q K^T (one 16x16 tile per wave)
    v8f sAcc;
    for (int v = 0; v < 8; ++v) sAcc[v] = 0.f;
    for (int kk = 0; kk < 128; kk += 32) {
      Frag aF, bF;
      const u16* ar = &Qs[om + cl][kk];
      aF.q[0] = *(const uint4*)(ar + ak);
      aF.q[1] = *(const uint4*)(ar + ak + 16);
      const u16* br = &Ks[sn + cl][kk];
      bF.q[0] = *(const uint4*)(br + bk);
      bF.q[1] = *(const uint4*)(br + bk + 16);
      sAcc = __builtin_amdgcn_wmma_f32_16x16x32_bf16(
          false, aF.v, false, bF.v, (short)0, sAcc, false, false);
    }
    for (int v = 0; v < 8; ++v)
      Ss[om + v + (hi ? 8 : 0)][sn + cl] = sAcc[v];
    __syncthreads();

    // online softmax (one thread per row)
    if (tid < 64) {
      int r = tid, grow = ib * 64 + r;
      float m0 = mS[r], mx = m0;
      for (int c = 0; c < 32; ++c) {
        int jc = jb * 32 + c;
        float sv = (jc <= grow) ? Ss[r][c] * scale : -1e30f;
        mx = fmaxf(mx, sv);
      }
      float alpha = __expf(m0 - mx);
      float sum = 0.f;
      for (int c = 0; c < 32; ++c) {
        int jc = jb * 32 + c;
        float p = 0.f;
        if (jc <= grow) { p = __expf(Ss[r][c] * scale - mx); sum += p; }
        Ps[r][c] = f2bf(p);
      }
      mS[r] = mx;
      lS[r] = lS[r] * alpha + sum;
      aS[r] = alpha;
    }
    __syncthreads();

    // rescale O, then O += P * V  (4 col tiles per wave, K=32)
    float av[8];
    for (int v = 0; v < 8; ++v) av[v] = aS[om + v + (hi ? 8 : 0)];
    for (int j = 0; j < 4; ++j)
      for (int v = 0; v < 8; ++v) o[j][v] *= av[v];
    {
      Frag aF;
      const u16* ar = &Ps[om + cl][0];
      aF.q[0] = *(const uint4*)(ar + ak);
      aF.q[1] = *(const uint4*)(ar + ak + 16);
      for (int j = 0; j < 4; ++j) {
        Frag bF;
        const u16* br = &Vt[on + j * 16 + cl][0];
        bF.q[0] = *(const uint4*)(br + bk);
        bF.q[1] = *(const uint4*)(br + bk + 16);
        o[j] = __builtin_amdgcn_wmma_f32_16x16x32_bf16(
            false, aF.v, false, bF.v, (short)0, o[j], false, false);
      }
    }
    __syncthreads();
  }

  // epilogue: divide by l, store bf16
  for (int j = 0; j < 4; ++j)
    for (int v = 0; v < 8; ++v) {
      int row = om + v + (hi ? 8 : 0);
      float val = o[j][v] / lS[row];
      int grow = ib * 64 + row;
      attn[(size_t)grow * 4096 + h * 128 + on + j * 16 + cl] = f2bf(val);
    }
}

// ---------------------------------------------------------------------------
// elementwise helpers
// ---------------------------------------------------------------------------
__global__ void silu_mul_kernel(const u16* __restrict__ g,
                                const u16* __restrict__ u,
                                u16* __restrict__ o, int n) {
  int i = blockIdx.x * 256 + threadIdx.x;
  if (i < n) {
    float gv = bf2f(g[i]);
    float s = gv / (1.f + __expf(-gv));
    o[i] = f2bf(s * bf2f(u[i]));
  }
}

__global__ void zero_f32_kernel(float* __restrict__ p, int n) {
  int i = blockIdx.x * 256 + threadIdx.x;
  if (i < n) p[i] = 0.f;
}

// ---------------------------------------------------------------------------
// host orchestration
// ---------------------------------------------------------------------------
extern "C" void kernel_launch(void* const* d_in, const int* in_sizes, int n_in,
                              void* d_out, int out_size, void* d_ws, size_t ws_size,
                              hipStream_t stream) {
  const float* x   = (const float*)d_in[0];
  const float* ln1 = (const float*)d_in[1];
  const float* ln2 = (const float*)d_in[2];
  const float* qUs = (const float*)d_in[3];
  const float* qV  = (const float*)d_in[4];
  const float* kUs = (const float*)d_in[5];
  const float* kV  = (const float*)d_in[6];
  const float* vUs = (const float*)d_in[7];
  const float* vV  = (const float*)d_in[8];
  const float* oUs = (const float*)d_in[9];
  const float* oV  = (const float*)d_in[10];
  const float* gUs = (const float*)d_in[11];
  const float* gV  = (const float*)d_in[12];
  const float* uUs = (const float*)d_in[13];
  const float* uV  = (const float*)d_in[14];
  const float* dUs = (const float*)d_in[15];
  const float* dV  = (const float*)d_in[16];
  float* out = (float*)d_out;

  char* w = (char*)d_ws;
  u16*   h1   = (u16*)(w + (0ull  << 20));  // 8 MB  [1024,4096]  (h1 / h2)
  u16*   xrq  = (u16*)(w + (8ull  << 20));  // 4 MB  [1024,2048]
  u16*   xrk  = (u16*)(w + (12ull << 20));  // 1 MB  [1024,512]
  u16*   xrv  = (u16*)(w + (13ull << 20));  // 1 MB  [1024,512]
  u16*   qb   = (u16*)(w + (14ull << 20));  // 8 MB  [1024,4096]
  u16*   kb   = (u16*)(w + (22ull << 20));  // 2 MB  [1024,1024]
  u16*   vb   = (u16*)(w + (24ull << 20));  // 2 MB  [1024,1024]
  u16*   attn = (u16*)(w + (26ull << 20));  // 8 MB  [1024,4096]
  u16*   orr  = (u16*)(w + (34ull << 20));  // 2 MB  [1024,1024]
  u16*   gr   = (u16*)(w + (36ull << 20));  // 2 MB  [1024,1024]
  u16*   ur   = (u16*)(w + (38ull << 20));  // 2 MB  [1024,1024]
  u16*   gch  = (u16*)(w + (40ull << 20));  // 4 MB  [1024,2048]
  u16*   uch  = (u16*)(w + (44ull << 20));  // 4 MB  [1024,2048]
  float* dr   = (float*)(w + (48ull << 20)); // 4 MB [1024,1024] fp32
  // total 52 MB of workspace

  // residual stream lives in d_out
  (void)hipMemcpyAsync(out, x, (size_t)1024 * 4096 * sizeof(float),
                       hipMemcpyDeviceToDevice, stream);

  // ---- attention half ----
  rmsnorm_kernel<<<1024, 256, 0, stream>>>(x, ln1, h1, 4096);

  // xr = h1 @ V^T (heads flattened into N)
  gemm_wmma_kernel<false, true, false><<<dim3(16, 8, 1), 256, 0, stream>>>(
      h1, 4096, 0, qV, 4096, 0, xrq, 2048, 0, 4096);
  gemm_wmma_kernel<false, true, false><<<dim3(4, 8, 1), 256, 0, stream>>>(
      h1, 4096, 0, kV, 4096, 0, xrk, 512, 0, 4096);
  gemm_wmma_kernel<false, true, false><<<dim3(4, 8, 1), 256, 0, stream>>>(
      h1, 4096, 0, vV, 4096, 0, xrv, 512, 0, 4096);

  // per-head q/k/v = xr @ Us^T (batched over heads via blockIdx.z)
  gemm_wmma_kernel<false, true, false><<<dim3(1, 8, 32), 256, 0, stream>>>(
      xrq, 2048, 64, qUs, 64, 128 * 64, qb, 4096, 128, 64);
  gemm_wmma_kernel<false, true, false><<<dim3(1, 8, 8), 256, 0, stream>>>(
      xrk, 512, 64, kUs, 64, 128 * 64, kb, 1024, 128, 64);
  gemm_wmma_kernel<false, true, false><<<dim3(1, 8, 8), 256, 0, stream>>>(
      xrv, 512, 64, vUs, 64, 128 * 64, vb, 1024, 128, 64);

  rope_kernel<<<1024, 256, 0, stream>>>(qb, kb);

  flash_kernel<<<dim3(16, 32), 256, 0, stream>>>(qb, kb, vb, attn);

  // low-rank O projection, accumulate into residual
  gemm_wmma_kernel<false, true, false><<<dim3(8, 8, 1), 256, 0, stream>>>(
      attn, 4096, 0, oV, 4096, 0, orr, 1024, 0, 4096);
  gemm_wmma_kernel<false, true, true><<<dim3(32, 8, 1), 256, 0, stream>>>(
      orr, 1024, 0, oUs, 1024, 0, out, 4096, 0, 1024);

  // ---- MLP half ----
  rmsnorm_kernel<<<1024, 256, 0, stream>>>(out, ln2, h1, 4096);  // h1 = h2

  gemm_wmma_kernel<false, true, false><<<dim3(8, 8, 1), 256, 0, stream>>>(
      h1, 4096, 0, gV, 4096, 0, gr, 1024, 0, 4096);
  gemm_wmma_kernel<false, true, false><<<dim3(8, 8, 1), 256, 0, stream>>>(
      h1, 4096, 0, uV, 4096, 0, ur, 1024, 0, 4096);

  zero_f32_kernel<<<4096, 256, 0, stream>>>(dr, 1024 * 1024);

  // chunk INTER=14336 into 7 x 2048 to bound workspace; dr accumulates in fp32
  for (int c = 0; c < 7; ++c) {
    const float* gU  = gUs + (size_t)c * 2048 * 1024;
    const float* uU  = uUs + (size_t)c * 2048 * 1024;
    const float* dVc = dV + (size_t)c * 2048;
    gemm_wmma_kernel<false, true, false><<<dim3(16, 8, 1), 256, 0, stream>>>(
        gr, 1024, 0, gU, 1024, 0, gch, 2048, 0, 1024);
    gemm_wmma_kernel<false, true, false><<<dim3(16, 8, 1), 256, 0, stream>>>(
        ur, 1024, 0, uU, 1024, 0, uch, 2048, 0, 1024);
    silu_mul_kernel<<<8192, 256, 0, stream>>>(gch, uch, gch, 1024 * 2048);
    gemm_wmma_kernel<false, true, true><<<dim3(8, 8, 1), 256, 0, stream>>>(
        gch, 2048, 0, dVc, 14336, 0, dr, 1024, 0, 2048);
  }

  // dn = dr @ d_Us^T, accumulate into residual
  gemm_wmma_kernel<true, true, true><<<dim3(32, 8, 1), 256, 0, stream>>>(
      dr, 1024, 0, dUs, 1024, 0, out, 4096, 0, 1024);
}
